// ScaNN_85048942395818
// MI455X (gfx1250) — matrix operand order repaired and checked by reading, not compiled, not run on hardware
//
#include <hip/hip_runtime.h>
#include <hip/hip_bf16.h>

// ---------------- problem constants (from reference) ----------------
#define BQ 128          // queries
#define DIM 128         // dims
#define JB 64           // codebooks
#define CB 16           // codes per book
#define NUM_LEAVES 100
#define L2SEARCH 10
#define NREORDER 100
#define TOPK 10
#define NEGV (-1e30f)

#define NBINS 2048
#define CAP 4096        // per-query collection capacity
#define TPB 8           // candidate tiles per block in the WMMA kernel

// ---------------- workspace layout (byte offsets) ----------------
// NOTE: QPK | CBPK | LEAF are contiguous on purpose: one TDM blob load covers all three.
#define OFF_QPK   0u                      // 128*64 u32  = 32768 B (queries packed bf16 pairs)
#define OFF_CBPK  32768u                  // 64*16 u32   = 4096 B (codebook packed bf16 pairs)
#define OFF_LEAF  36864u                  // 128*4 u32   = 2048 B (leaf membership bitmasks)
#define STATIC_DW (BQ*JB + JB*CB + BQ*4)  // 9728 dwords = 38912 B
#define OFF_THR   38912u                  // 128 u32
#define OFF_CNT   39424u                  // 128 u32
#define OFF_SELS  39936u                  // 128*100 f32
#define OFF_SELI  91136u                  // 128*100 i32
#define OFF_HIST  142336u                 // 128*2048 u32 = 1 MB
#define OFF_COLL  1190912u                // 128*4096*2 u32 = 4 MB

typedef __attribute__((ext_vector_type(16))) __bf16 v16bf;
typedef __attribute__((ext_vector_type(8)))  float  v8f;
typedef __attribute__((ext_vector_type(4)))  unsigned v4u;
typedef __attribute__((ext_vector_type(8)))  int    v8i;
typedef __attribute__((ext_vector_type(4)))  int    v4i;

union Frag { unsigned u[8]; v16bf v; };

__device__ __forceinline__ unsigned short f2bf(float f) {
    unsigned u = __float_as_uint(f);
    unsigned r = (u + 0x7FFFu + ((u >> 16) & 1u)) >> 16;   // RNE
    return (unsigned short)r;
}
__device__ __forceinline__ unsigned pack2bf(float lo, float hi) {
    return (unsigned)f2bf(lo) | ((unsigned)f2bf(hi) << 16);
}
__device__ __forceinline__ unsigned monou(unsigned u) {
    return (u & 0x80000000u) ? ~u : (u | 0x80000000u);      // monotonic float->uint
}
__device__ __forceinline__ unsigned monof(float f) { return monou(__float_as_uint(f)); }

// ---- Tensor Data Mover: 2D tile (widths in dwords) from global to LDS ----
// pad64: insert 1 dword of LDS padding after every 64 dwords stored (row stride 65)
__device__ __forceinline__ void tdm_load_2d(unsigned ldsOff, const void* gptr,
                                            unsigned tensorW, unsigned tensorH,
                                            unsigned tileW, unsigned tileH,
                                            unsigned strideW, bool pad64) {
    unsigned long long ga = (unsigned long long)(size_t)gptr;
    v4u g0;
    g0.x = 1u;                                              // count=1, user descriptor
    g0.y = ldsOff;                                          // lds_addr (bytes)
    g0.z = (unsigned)(ga & 0xFFFFFFFFu);                    // global_addr[31:0]
    g0.w = (unsigned)((ga >> 32) & 0x1FFFFFFu) | 0x80000000u; // global_addr[56:32] | type=2
    v8i g1;
    g1[0] = (int)((2u << 16) | (pad64 ? ((1u << 20) | (5u << 22)) : 0u)); // data_size=4B (+pad ctl)
    g1[1] = (int)((tensorW & 0xFFFFu) << 16);               // tensor_dim0[15:0]
    g1[2] = (int)((tensorW >> 16) | ((tensorH & 0xFFFFu) << 16)); // td0 hi | tensor_dim1 lo
    g1[3] = (int)((tensorH >> 16) | (tileW << 16));         // td1 hi | tile_dim0
    g1[4] = (int)tileH;                                     // tile_dim1 (tile_dim2=0)
    g1[5] = (int)strideW;                                   // tensor_dim0_stride lo
    g1[6] = 0; g1[7] = 0;
    v4i gz = {0, 0, 0, 0};
#if __clang_major__ >= 23
    v8i gz8 = {0, 0, 0, 0, 0, 0, 0, 0};
    __builtin_amdgcn_tensor_load_to_lds(g0, g1, gz, gz, gz8, 0);
#else
    __builtin_amdgcn_tensor_load_to_lds(g0, g1, gz, gz, 0);
#endif
}

// ---------------- kernel 0: pack queries/codebooks to bf16 pairs, zero hist/cnt ----------------
__global__ __launch_bounds__(256) void k_prep(const float* __restrict__ queries,
                                              const float* __restrict__ codebooks,
                                              unsigned* __restrict__ qpk,
                                              unsigned* __restrict__ cbpk,
                                              unsigned* __restrict__ hist,
                                              unsigned* __restrict__ cnt) {
    int idx = blockIdx.x * blockDim.x + threadIdx.x;
    int total = gridDim.x * blockDim.x;
    if (idx < BQ * JB) {
        int b = idx >> 6, j = idx & 63;
        qpk[idx] = pack2bf(queries[b * DIM + 2 * j], queries[b * DIM + 2 * j + 1]);
    }
    if (idx < JB * CB) {  // codebooks[j][c][0..1] -> one packed bf16 pair
        cbpk[idx] = pack2bf(codebooks[idx * 2 + 0], codebooks[idx * 2 + 1]);
    }
    for (int i = idx; i < BQ * NBINS; i += total) hist[i] = 0u;
    if (idx < BQ) cnt[idx] = 0u;
}

// ---------------- kernel 1: center scores + top-10 leaves -> bitmask ----------------
__global__ __launch_bounds__(128) void k_centers(const float* __restrict__ queries,
                                                 const float* __restrict__ centers,
                                                 unsigned* __restrict__ leafmask) {
    int q = blockIdx.x, t = threadIdx.x;
    __shared__ float sq[DIM];
    __shared__ float sc[NUM_LEAVES];
    sq[t] = queries[q * DIM + t];
    __syncthreads();
    if (t < NUM_LEAVES) {
        float acc = 0.f;
        for (int d = 0; d < DIM; ++d) acc += sq[d] * centers[t * DIM + d];
        sc[t] = acc;
    }
    __syncthreads();
    if (t == 0) {
        unsigned m[4] = {0u, 0u, 0u, 0u};
        for (int i = 0; i < L2SEARCH; ++i) {
            float best = -3.4e38f; int bi = 0;
            for (int c = 0; c < NUM_LEAVES; ++c)
                if (sc[c] > best) { best = sc[c]; bi = c; }
            sc[bi] = -3.4e38f;
            m[bi >> 5] |= 1u << (bi & 31);
        }
        leafmask[q * 4 + 0] = m[0]; leafmask[q * 4 + 1] = m[1];
        leafmask[q * 4 + 2] = m[2]; leafmask[q * 4 + 3] = m[3];
    }
}

// ---------------- kernel 2: WMMA AH scoring; PASS 0 = histogram, PASS 1 = collect ----------------
// Block = 256 threads = 8 waves; wave w owns query tile w (16 queries).
// Each block walks TPB candidate tiles; codes tiles stream via double-buffered TDM.
template <int PASS>
__global__ __launch_bounds__(256) void k_score(const int* __restrict__ codes,
                                               const int* __restrict__ cand_leaf,
                                               const unsigned* __restrict__ qpk,   // base of static blob
                                               const unsigned* __restrict__ thresh,
                                               unsigned* __restrict__ hist,
                                               unsigned* __restrict__ cnt,
                                               unsigned* __restrict__ coll,
                                               int nCand, int nTiles) {
    __shared__ __align__(16) unsigned s_static[STATIC_DW]; // qpk | cbpk | leafmask (one TDM blob)
    __shared__ __align__(16) int s_codes[2][16 * 65];      // padded rows: stride 65 dwords
    __shared__ int s_leaf[16];

    unsigned* s_qpk  = s_static;
    unsigned* s_cbpk = s_static + BQ * JB;
    unsigned* s_lm   = s_static + BQ * JB + JB * CB;

    const int t = threadIdx.x;
    const int tileBase = blockIdx.x * TPB;

    // Prologue: wave 0 launches async TDM for the static blob + first codes tile.
    if (t < 32) {
        tdm_load_2d((unsigned)(size_t)(void*)s_static, qpk,
                    STATIC_DW, 1, STATIC_DW, 1, STATIC_DW, false);
        int rem = nCand - tileBase * 16;
        tdm_load_2d((unsigned)(size_t)(void*)&s_codes[0][0], codes + (size_t)tileBase * 16 * JB,
                    JB, (unsigned)(rem < 16 ? rem : 16), JB, 16, JB, true);
    }

    const int wv = t >> 5;          // query tile 0..7
    const int lane = t & 31;
    const int h = lane >> 4;        // half-wave
    const int mn = lane & 15;       // A: row M, B: col N

    for (int it = 0; it < TPB; ++it) {
        const int ct = tileBase + it;
        // Issue next tile, then wait so the CURRENT tile is resident (TDM completes in order).
        if (t < 32) {
            int nt = ct + 1;
            if (it + 1 < TPB && nt < nTiles) {
                int rem = nCand - nt * 16;
                tdm_load_2d((unsigned)(size_t)(void*)&s_codes[(it + 1) & 1][0],
                            codes + (size_t)nt * 16 * JB,
                            JB, (unsigned)(rem < 16 ? rem : 16), JB, 16, JB, true);
                __builtin_amdgcn_s_wait_tensorcnt(1);
            } else {
                __builtin_amdgcn_s_wait_tensorcnt(0);
            }
        }
        if (ct < nTiles && t < 16) {
            int cg = ct * 16 + t;
            s_leaf[t] = (cg < nCand) ? cand_leaf[cg] : 0;
        }
        if (t == 0 && ct + 1 < nTiles)
            __builtin_prefetch(cand_leaf + (size_t)(ct + 1) * 16, 0, 1); // global_prefetch_b8
        __syncthreads();

        if (ct < nTiles) {
            const int* cbuf = &s_codes[it & 1][0];
            v8f acc = {0.f, 0.f, 0.f, 0.f, 0.f, 0.f, 0.f, 0.f};
#pragma unroll
            for (int s = 0; s < 4; ++s) {   // K = 128 = 4 x 32
                Frag a, b;
#pragma unroll
                for (int v = 0; v < 8; ++v) {
                    // A 16x32 bf16 layout: K-pair j = h*8 + v (+16 per step)
                    int ja = s * 16 + h * 8 + v;
                    a.u[v] = s_qpk[(wv * 16 + mn) * JB + ja];
                    // B 32x16 bf16 layout: j = h*8 + v (+16 per step)
                    int jb = s * 16 + h * 8 + v;
                    int c = cbuf[mn * 65 + jb];         // TDM-padded rows: conflict-free
                    b.u[v] = s_cbpk[(jb << 4) + c];     // reconstructed pair, bf16x2
                }
                acc = __builtin_amdgcn_wmma_f32_16x16x32_bf16(false, a.v, false, b.v,
                                                              (short)0, acc, false, false);
            }

            // C layout: lane (h,mn), vgpr r -> score[q = wv*16 + r + 8h][cand = ct*16 + mn]
            const int cg = ct * 16 + mn;
            if (cg < nCand) {
                const int leaf = s_leaf[mn];
#pragma unroll
                for (int r = 0; r < 8; ++r) {
                    int q = wv * 16 + r + 8 * h;
                    bool member = (s_lm[q * 4 + (leaf >> 5)] >> (leaf & 31)) & 1u;
                    if (!member) continue;
                    float sc = acc[r];
                    unsigned bin = monof(sc) >> 21;     // 2048 bins
                    if (PASS == 0) {
                        atomicAdd(&hist[q * NBINS + bin], 1u);
                    } else if (bin >= thresh[q]) {
                        unsigned pos = atomicAdd(&cnt[q], 1u);
                        if (pos < CAP) {
                            coll[((size_t)q * CAP + pos) * 2 + 0] = __float_as_uint(sc);
                            coll[((size_t)q * CAP + pos) * 2 + 1] = (unsigned)cg;
                        }
                    }
                }
            }
        }
        __syncthreads();
    }
}

// ---------------- kernel 2b: per-query threshold bin from histogram ----------------
__global__ void k_thresh(const unsigned* __restrict__ hist,
                         unsigned* __restrict__ thresh,
                         unsigned* __restrict__ cnt) {
    int q = threadIdx.x;
    if (q < BQ) {
        unsigned cum = 0, tb = 0;
        for (int b = NBINS - 1; b >= 0; --b) {
            cum += hist[q * NBINS + b];
            if (cum >= NREORDER) { tb = (unsigned)b; break; }
        }
        thresh[q] = tb;
        cnt[q] = 0u;    // reset for collection pass
    }
}

// ---------------- kernel 3: per-query exact top-100 of collected set ----------------
__global__ __launch_bounds__(256) void k_sel(const unsigned* __restrict__ coll,
                                             const unsigned* __restrict__ cnt,
                                             float* __restrict__ sel_s,
                                             int* __restrict__ sel_i) {
    int q = blockIdx.x, t = threadIdx.x;
    __shared__ unsigned long long key[CAP];   // 32 KB
    __shared__ unsigned long long red[256];
    unsigned cc = cnt[q];
    int c = (cc < (unsigned)CAP) ? (int)cc : CAP;
    for (int i = t; i < CAP; i += 256) {
        unsigned long long k = 0ull;
        if (i < c) {
            unsigned sb = coll[((size_t)q * CAP + i) * 2 + 0];
            unsigned id = coll[((size_t)q * CAP + i) * 2 + 1];
            // max key = max score; ties -> smallest id (matches top_k first-index)
            k = ((unsigned long long)monou(sb) << 32) | (unsigned long long)(0xFFFFFFFFu - id);
        }
        key[i] = k;
    }
    __syncthreads();
    for (int r = 0; r < NREORDER; ++r) {
        unsigned long long best = 0ull;
        for (int i = t; i < c; i += 256) best = (key[i] > best) ? key[i] : best;
        red[t] = best;
        __syncthreads();
        for (int s = 128; s > 0; s >>= 1) {
            if (t < s) red[t] = (red[t + s] > red[t]) ? red[t + s] : red[t];
            __syncthreads();
        }
        unsigned long long win = red[0];
        if (win != 0ull) {
            for (int i = t; i < c; i += 256)
                if (key[i] == win) key[i] = 0ull;
            if (t == 0) {
                unsigned m = (unsigned)(win >> 32);
                unsigned ub = (m & 0x80000000u) ? (m ^ 0x80000000u) : ~m;
                sel_s[q * NREORDER + r] = __uint_as_float(ub);
                sel_i[q * NREORDER + r] = (int)(0xFFFFFFFFu - (unsigned)(win & 0xFFFFFFFFull));
            }
        } else if (t == 0) {
            sel_s[q * NREORDER + r] = NEGV;
            sel_i[q * NREORDER + r] = 0;
        }
        __syncthreads();
    }
}

// ---------------- kernel 4: exact f32 rescore + top-10 ----------------
__global__ __launch_bounds__(128) void k_exact(const float* __restrict__ queries,
                                               const float* __restrict__ candidates,
                                               const int* __restrict__ identifiers,
                                               const float* __restrict__ sel_s,
                                               const int* __restrict__ sel_i,
                                               float* __restrict__ out) {
    int q = blockIdx.x, t = threadIdx.x;
    int wv = t >> 5, lane = t & 31;
    __shared__ float ex[NREORDER];
    __shared__ int eid[NREORDER];
    for (int r = wv; r < NREORDER; r += 4) {
        int cand = sel_i[q * NREORDER + r];
        float ah = sel_s[q * NREORDER + r];
        float p = 0.f;
        for (int d = lane; d < DIM; d += 32)
            p += queries[q * DIM + d] * candidates[(size_t)cand * DIM + d];
        for (int o = 16; o > 0; o >>= 1) p += __shfl_down(p, o, 32);
        if (lane == 0) { ex[r] = (ah > NEGV * 0.5f) ? p : NEGV; eid[r] = cand; }
    }
    __syncthreads();
    if (t == 0) {
        for (int i = 0; i < TOPK; ++i) {
            float best = -3.4e38f; int bi = 0;
            for (int r = 0; r < NREORDER; ++r)
                if (ex[r] > best) { best = ex[r]; bi = r; }
            out[q * TOPK + i] = best;
            out[BQ * TOPK + q * TOPK + i] = (float)identifiers[eid[bi]];
            ex[bi] = -3.4e38f;
        }
    }
}

// ---------------- host launcher ----------------
extern "C" void kernel_launch(void* const* d_in, const int* in_sizes, int n_in,
                              void* d_out, int out_size, void* d_ws, size_t ws_size,
                              hipStream_t stream) {
    const float* queries    = (const float*)d_in[0];
    const float* candidates = (const float*)d_in[1];
    const float* centers    = (const float*)d_in[2];
    const float* codebooks  = (const float*)d_in[3];
    const int*   codes      = (const int*)d_in[4];
    const int*   cand_leaf  = (const int*)d_in[5];
    const int*   identifiers= (const int*)d_in[6];

    const int N = in_sizes[5];              // candidates count
    const int nTiles = (N + 15) / 16;
    const int nBlocks = (nTiles + TPB - 1) / TPB;

    char* ws = (char*)d_ws;
    unsigned* qpk    = (unsigned*)(ws + OFF_QPK);
    unsigned* cbpk   = (unsigned*)(ws + OFF_CBPK);
    unsigned* lmask  = (unsigned*)(ws + OFF_LEAF);
    unsigned* thr    = (unsigned*)(ws + OFF_THR);
    unsigned* cnt    = (unsigned*)(ws + OFF_CNT);
    float*    sel_s  = (float*)   (ws + OFF_SELS);
    int*      sel_i  = (int*)     (ws + OFF_SELI);
    unsigned* hist   = (unsigned*)(ws + OFF_HIST);
    unsigned* coll   = (unsigned*)(ws + OFF_COLL);
    float* out = (float*)d_out;

    k_prep<<<512, 256, 0, stream>>>(queries, codebooks, qpk, cbpk, hist, cnt);
    k_centers<<<BQ, 128, 0, stream>>>(queries, centers, lmask);
    k_score<0><<<nBlocks, 256, 0, stream>>>(codes, cand_leaf, qpk, thr,
                                            hist, cnt, coll, N, nTiles);
    k_thresh<<<1, 128, 0, stream>>>(hist, thr, cnt);
    k_score<1><<<nBlocks, 256, 0, stream>>>(codes, cand_leaf, qpk, thr,
                                            hist, cnt, coll, N, nTiles);
    k_sel<<<BQ, 256, 0, stream>>>(coll, cnt, sel_s, sel_i);
    k_exact<<<BQ, 128, 0, stream>>>(queries, candidates, identifiers, sel_s, sel_i, out);
}
